// MultiHeadAttentionLayer_47090021433672
// MI455X (gfx1250) — matrix operand order repaired and checked by reading, not compiled, not run on hardware
//
#include <hip/hip_runtime.h>

// MultiHeadAttention (B=4, L=2048, E=512, H=8, D=64) for MI455X / gfx1250.
// ~51.6 GFLOP vs ~70MB traffic -> compute bound; use v_wmma_f32_16x16x32_f16
// (f16 in / f32 acc) for all four GEMM stages, flash-attention style softmax.

#define E     512
#define HEADS 8
#define HD    64
#define BATCH 4
#define SEQ   2048
#define NTOK  (BATCH * SEQ)

typedef __attribute__((ext_vector_type(16))) _Float16 v16h;
typedef __attribute__((ext_vector_type(8)))  _Float16 v8h;
typedef __attribute__((ext_vector_type(8)))  float    v8f;

// Build a 16-half WMMA fragment from two 16-byte chunks.
__device__ __forceinline__ v16h load_frag2(const _Float16* p0, const _Float16* p1) {
  union { v16h v; v8h h[2]; } u;
  u.h[0] = *(const v8h*)p0;
  u.h[1] = *(const v8h*)p1;
  return u.v;
}

__device__ __forceinline__ v8f wmma_f16(v16h a, v16h b, v8f c) {
  return __builtin_amdgcn_wmma_f32_16x16x32_f16(false, a, false, b, (short)0, c,
                                                false, false);
}

// ---------------------------------------------------------------- conversion
__global__ void cvt_f32_to_f16(const float* __restrict__ src,
                               _Float16* __restrict__ dst, int n) {
  int i = blockIdx.x * blockDim.x + threadIdx.x;
  if (i < n) dst[i] = (_Float16)src[i];
}

// ------------------------------------------------------- QKV projection GEMM
// Y = X @ W^T + bias.  X:[NTOK,E] f16, W:[E,E] f16 (row = out feature).
// transposed==0 : dst is [B,H,L,D]    (Q and K)
// transposed==1 : dst is [B,H,D,L]    (V stored pre-transposed for P@V)
__global__ __launch_bounds__(256)
void qkv_proj(const _Float16* __restrict__ X, const _Float16* __restrict__ W,
              const float* __restrict__ bias, _Float16* __restrict__ dst,
              int transposed) {
  const int lane  = threadIdx.x & 31;
  const int wave  = threadIdx.x >> 5;     // 8 waves: each owns 64 of 512 N
  const int ln15  = lane & 15;
  const int hi    = lane >> 4;
  const int mtile = blockIdx.x * 16;
  const int nbase = wave * 64;

  v8f acc[4] = {};
  const _Float16* ap = X + (size_t)(mtile + ln15) * E + hi * 16;

  for (int ks = 0; ks < E; ks += 32) {
    v16h a = load_frag2(ap + ks, ap + ks + 8);
#pragma unroll
    for (int nt = 0; nt < 4; ++nt) {
      const _Float16* bp =
          W + (size_t)(nbase + nt * 16 + ln15) * E + ks + hi * 16;
      v16h b = load_frag2(bp, bp + 8);
      acc[nt] = wmma_f16(a, b, acc[nt]);
    }
  }

#pragma unroll
  for (int nt = 0; nt < 4; ++nt) {
    const int n  = nbase + nt * 16 + ln15;
    const int h  = n >> 6;            // / HD
    const int d  = n & 63;
    const float bv = bias[n];
#pragma unroll
    for (int j = 0; j < 8; ++j) {
      const int m = mtile + j + hi * 8;    // global token row
      const int b = m >> 11;               // / SEQ
      const int l = m & 2047;
      const float v = acc[nt][j] + bv;
      size_t off = transposed
          ? ((size_t)((b * HEADS + h) * HD + d)) * SEQ + l
          : ((size_t)((b * HEADS + h) * SEQ + l)) * HD + d;
      dst[off] = (_Float16)v;
    }
  }
}

// ------------------------------------------------------------ flash attention
// Q,K : [B,H,L,D] f16   VT : [B,H,D,L] f16   O : [B,L,E] f16
// Block = 128 threads (4 waves); each wave owns a 16-query tile and streams
// the 2048 keys in chunks of 32 with online softmax.
__global__ __launch_bounds__(128)
void flash_attn(const _Float16* __restrict__ Q, const _Float16* __restrict__ K,
                const _Float16* __restrict__ VT, _Float16* __restrict__ O) {
  __shared__ __align__(16) _Float16 Pl[4][16 * 32];  // per-wave P staging

  const int lane  = threadIdx.x & 31;
  const int wave  = threadIdx.x >> 5;
  const int ln15  = lane & 15;
  const int hi    = lane >> 4;
  const int bh    = blockIdx.y;            // b*HEADS + h
  const int b     = bh >> 3;
  const int h     = bh & 7;
  const int qtile = (blockIdx.x * 4 + wave) * 16;

  const _Float16* Qb = Q  + (size_t)bh * SEQ * HD;
  const _Float16* Kb = K  + (size_t)bh * SEQ * HD;
  const _Float16* Vb = VT + (size_t)bh * HD * SEQ;

  // Q A-fragments for d=[0,32) and [32,64), kept in registers whole loop.
  const _Float16* qr = Qb + (size_t)(qtile + ln15) * HD + hi * 16;
  const v16h aQ0 = load_frag2(qr,      qr + 8);
  const v16h aQ1 = load_frag2(qr + 32, qr + 40);

  float mrow[8], lrow[8];
#pragma unroll
  for (int j = 0; j < 8; ++j) { mrow[j] = -1e30f; lrow[j] = 0.f; }
  v8f o[4] = {};

  const float L2E = 1.44269504088896340736f;

  for (int kt = 0; kt < SEQ; kt += 32) {
    // ---- scores = Q @ K^T for 32 keys (2 N-tiles x 2 K-chunks = 4 WMMA)
    v8f sc[2];
#pragma unroll
    for (int t = 0; t < 2; ++t) {
      const _Float16* kr =
          Kb + (size_t)(kt + t * 16 + ln15) * HD + hi * 16;
      v16h b0 = load_frag2(kr,      kr + 8);
      v16h b1 = load_frag2(kr + 32, kr + 40);
      v8f c = {};
      c = wmma_f16(aQ0, b0, c);
      c = wmma_f16(aQ1, b1, c);
      sc[t] = c;
    }

    // ---- online softmax (rows live across 16-lane halves in C layout)
    float scale[8];
#pragma unroll
    for (int j = 0; j < 8; ++j) {
      float s0 = sc[0][j] * 0.125f;   // 1/sqrt(64)
      float s1 = sc[1][j] * 0.125f;
      float t = fmaxf(s0, s1);
      t = fmaxf(t, __shfl_xor(t, 8, 16));
      t = fmaxf(t, __shfl_xor(t, 4, 16));
      t = fmaxf(t, __shfl_xor(t, 2, 16));
      t = fmaxf(t, __shfl_xor(t, 1, 16));
      const float mnew = fmaxf(mrow[j], t);
      const float sf   = exp2f((mrow[j] - mnew) * L2E);
      const float p0   = exp2f((s0 - mnew) * L2E);
      const float p1   = exp2f((s1 - mnew) * L2E);
      float rs = p0 + p1;
      rs += __shfl_xor(rs, 8, 16);
      rs += __shfl_xor(rs, 4, 16);
      rs += __shfl_xor(rs, 2, 16);
      rs += __shfl_xor(rs, 1, 16);
      lrow[j] = lrow[j] * sf + rs;
      mrow[j] = mnew;
      scale[j] = sf;
      // stash P (C layout -> row-major 16x32 tile in LDS)
      const int m = j + hi * 8;
      Pl[wave][m * 32 + ln15]      = (_Float16)p0;
      Pl[wave][m * 32 + 16 + ln15] = (_Float16)p1;
    }

    // rescale running output by per-row correction
#pragma unroll
    for (int nt = 0; nt < 4; ++nt)
#pragma unroll
      for (int j = 0; j < 8; ++j) o[nt][j] *= scale[j];

    // ---- P (as A-frag from LDS) @ V  (4 WMMA across d)
    const _Float16* pp = &Pl[wave][ln15 * 32 + hi * 16];
    const v16h aP = load_frag2(pp, pp + 8);
#pragma unroll
    for (int nt = 0; nt < 4; ++nt) {
      const _Float16* vr =
          Vb + (size_t)(nt * 16 + ln15) * SEQ + kt + hi * 16;
      v16h bV = load_frag2(vr, vr + 8);
      o[nt] = wmma_f16(aP, bV, o[nt]);
    }
  }

  // ---- normalize and emit O[b, q, h*64 + d] in f16
#pragma unroll
  for (int j = 0; j < 8; ++j) {
    const float inv = 1.f / lrow[j];
    const int m = qtile + j + hi * 8;
    const size_t base = ((size_t)(b * SEQ + m)) * E + h * HD;
#pragma unroll
    for (int nt = 0; nt < 4; ++nt)
      O[base + nt * 16 + ln15] = (_Float16)(o[nt][j] * inv);
  }
}

// ---------------------------------- out-projection + bias + residual + LN
__global__ __launch_bounds__(256)
void out_proj_ln(const _Float16* __restrict__ O16,
                 const _Float16* __restrict__ Wo, const float* __restrict__ bo,
                 const float* __restrict__ resid,  // original f32 query
                 const float* __restrict__ g, const float* __restrict__ beta,
                 float* __restrict__ out) {
  __shared__ float red[8][16][2];
  __shared__ float musig[16][2];

  const int lane  = threadIdx.x & 31;
  const int wave  = threadIdx.x >> 5;
  const int ln15  = lane & 15;
  const int hi    = lane >> 4;
  const int mtile = blockIdx.x * 16;
  const int nbase = wave * 64;

  v8f acc[4] = {};
  const _Float16* ap = O16 + (size_t)(mtile + ln15) * E + hi * 16;
  for (int ks = 0; ks < E; ks += 32) {
    v16h a = load_frag2(ap + ks, ap + ks + 8);
#pragma unroll
    for (int nt = 0; nt < 4; ++nt) {
      const _Float16* bp =
          Wo + (size_t)(nbase + nt * 16 + ln15) * E + ks + hi * 16;
      v16h bb = load_frag2(bp, bp + 8);
      acc[nt] = wmma_f16(a, bb, acc[nt]);
    }
  }

  // x = gemm + bias + residual; per-row partial moments
  float s1[8] = {}, s2[8] = {};
#pragma unroll
  for (int nt = 0; nt < 4; ++nt) {
    const int n = nbase + nt * 16 + ln15;
    const float bv = bo[n];
#pragma unroll
    for (int j = 0; j < 8; ++j) {
      const int m = mtile + j + hi * 8;
      float x = acc[nt][j] + bv + resid[(size_t)m * E + n];
      acc[nt][j] = x;
      s1[j] += x;
      s2[j] += x * x;
    }
  }
#pragma unroll
  for (int j = 0; j < 8; ++j) {
    for (int off = 8; off >= 1; off >>= 1) {
      s1[j] += __shfl_xor(s1[j], off, 16);
      s2[j] += __shfl_xor(s2[j], off, 16);
    }
  }
  if (ln15 == 0) {
#pragma unroll
    for (int j = 0; j < 8; ++j) {
      red[wave][hi * 8 + j][0] = s1[j];
      red[wave][hi * 8 + j][1] = s2[j];
    }
  }
  __syncthreads();
  if (threadIdx.x < 16) {
    float a = 0.f, b = 0.f;
    for (int w = 0; w < 8; ++w) {
      a += red[w][threadIdx.x][0];
      b += red[w][threadIdx.x][1];
    }
    const float mu  = a * (1.f / E);
    const float var = b * (1.f / E) - mu * mu;
    musig[threadIdx.x][0] = mu;
    musig[threadIdx.x][1] = rsqrtf(var + 1e-5f);
  }
  __syncthreads();

#pragma unroll
  for (int j = 0; j < 8; ++j) {
    const int m  = mtile + j + hi * 8;
    const float mu = musig[hi * 8 + j][0];
    const float rs = musig[hi * 8 + j][1];
#pragma unroll
    for (int nt = 0; nt < 4; ++nt) {
      const int n = nbase + nt * 16 + ln15;
      out[(size_t)m * E + n] = (acc[nt][j] - mu) * rs * g[n] + beta[n];
    }
  }
}

// --------------------------------------------------------------------- launch
extern "C" void kernel_launch(void* const* d_in, const int* in_sizes, int n_in,
                              void* d_out, int out_size, void* d_ws,
                              size_t ws_size, hipStream_t stream) {
  const float* values = (const float*)d_in[0];
  const float* keys   = (const float*)d_in[1];
  const float* query  = (const float*)d_in[2];
  const float* Wq = (const float*)d_in[3];
  const float* bq = (const float*)d_in[4];
  const float* Wk = (const float*)d_in[5];
  const float* bk = (const float*)d_in[6];
  const float* Wv = (const float*)d_in[7];
  const float* bv = (const float*)d_in[8];
  const float* Wo = (const float*)d_in[9];
  const float* bo = (const float*)d_in[10];
  const float* ln_g = (const float*)d_in[11];
  const float* ln_b = (const float*)d_in[12];
  float* out = (float*)d_out;

  const size_t TOKE = (size_t)NTOK * E;  // 4,194,304
  const size_t WE   = (size_t)E * E;     // 262,144
  _Float16* ws   = (_Float16*)d_ws;      // total ~58 MB of scratch used
  _Float16* xq   = ws;
  _Float16* xk   = xq + TOKE;
  _Float16* xv   = xk + TOKE;
  _Float16* wq16 = xv + TOKE;
  _Float16* wk16 = wq16 + WE;
  _Float16* wv16 = wk16 + WE;
  _Float16* wo16 = wv16 + WE;
  _Float16* Qh   = wo16 + WE;
  _Float16* Kh   = Qh + TOKE;
  _Float16* VTh  = Kh + TOKE;
  _Float16* O16  = VTh + TOKE;

  const int thr = 256;
  cvt_f32_to_f16<<<(int)(TOKE / thr), thr, 0, stream>>>(query,  xq, (int)TOKE);
  cvt_f32_to_f16<<<(int)(TOKE / thr), thr, 0, stream>>>(keys,   xk, (int)TOKE);
  cvt_f32_to_f16<<<(int)(TOKE / thr), thr, 0, stream>>>(values, xv, (int)TOKE);
  cvt_f32_to_f16<<<(int)(WE / thr),   thr, 0, stream>>>(Wq, wq16, (int)WE);
  cvt_f32_to_f16<<<(int)(WE / thr),   thr, 0, stream>>>(Wk, wk16, (int)WE);
  cvt_f32_to_f16<<<(int)(WE / thr),   thr, 0, stream>>>(Wv, wv16, (int)WE);
  cvt_f32_to_f16<<<(int)(WE / thr),   thr, 0, stream>>>(Wo, wo16, (int)WE);

  qkv_proj<<<NTOK / 16, 256, 0, stream>>>(xq, wq16, bq, Qh, 0);
  qkv_proj<<<NTOK / 16, 256, 0, stream>>>(xk, wk16, bk, Kh, 0);
  qkv_proj<<<NTOK / 16, 256, 0, stream>>>(xv, wv16, bv, VTh, 1);

  flash_attn<<<dim3(SEQ / 64, BATCH * HEADS), 128, 0, stream>>>(Qh, Kh, VTh,
                                                                O16);

  out_proj_ln<<<NTOK / 16, 256, 0, stream>>>(O16, wo16, bo, query, ln_g, ln_b,
                                             out);
}
